// GGNNRelReason_13357348291348
// MI455X (gfx1250) — matrix operand order repaired and checked by reading, not compiled, not run on hardware
//
#include <hip/hip_runtime.h>
#include <hip/hip_bf16.h>

typedef __attribute__((ext_vector_type(16))) _Float16 v16h;
typedef __attribute__((ext_vector_type(8)))  float    v8f;

#define BM 128
#define BN 64
#define BK 32
#define LDS_STRIDE (BK + 4)   // halfs; even -> u32-aligned pair reads

// ---------------------------------------------------------------------------
// Generic WMMA GEMM: C[M,N] = op( A'[M,K] @ B[K,N] (+beta*C) (+bias) ),
// where A'[m,:] = A[gather[m],:] if gather != nullptr.
// A,B,C row-major f32; compute in f16 fragments, f32 WMMA accumulation.
// Block 256 = 8 waves (4x2), tile 128x64x32; each wave does 4 WMMAs/K-step.
// EXACT=true: M%BM==0, N%BN==0, K%BK==0, A/B 16B-aligned, lda/ldb%4==0
//             -> completely guard-free staging and epilogue.
// ---------------------------------------------------------------------------
template <bool EXACT>
__global__ void __launch_bounds__(256)
gemm_wmma_kernel(const float* __restrict__ A, int lda,
                 const float* __restrict__ B, int ldb,
                 float* __restrict__ C, int ldc,
                 int M, int N, int K,
                 const float* __restrict__ bias,
                 const int* __restrict__ gather,
                 int beta, int relu)
{
    __shared__ _Float16 As[BM][LDS_STRIDE];
    __shared__ _Float16 Bs[BN][LDS_STRIDE];   // B staged transposed: Bs[n][k]

    const int tid  = threadIdx.x;
    const int lane = tid & 31;
    const int wid  = tid >> 5;       // 0..7
    const int wm   = wid >> 1;       // 0..3 -> 32-row strip
    const int wn   = wid & 1;        // 0..1 -> 32-col strip
    const int hi   = lane >> 4;      // lane half
    const int l15  = lane & 15;

    const int n0 = blockIdx.x * BN;
    const int m0 = blockIdx.y * BM;

    // ---- per-thread staging ownership (constant across K loop) ----
    const int arow_l = tid >> 1;           // 0..127
    const int akoff  = (tid & 1) * 16;     // 0 or 16
    const int agm    = m0 + arow_l;
    const bool arow_ok = EXACT || (agm < M);
    const float* Arow = A;
    if (arow_ok) {
        const int ar = gather ? gather[agm] : agm;   // hoisted: one load total
        Arow = A + (long)ar * lda;
    }
    const int bkk = tid >> 3;              // 0..31
    const int bnb = (tid & 7) * 8;         // 0..56

    v8f acc[2][2] = {};

    for (int k0 = 0; k0 < K; k0 += BK) {
        // ---- stage A tile (f32 -> f16) ----
        if (EXACT) {
            const float4* src = (const float4*)(Arow + k0 + akoff);
            #pragma unroll
            for (int q = 0; q < 4; ++q) {
                float4 v = src[q];
                union { _Float16 h[2]; unsigned u; } p0, p1;
                p0.h[0] = (_Float16)v.x; p0.h[1] = (_Float16)v.y;
                p1.h[0] = (_Float16)v.z; p1.h[1] = (_Float16)v.w;
                *(unsigned*)&As[arow_l][akoff + 4 * q]     = p0.u;
                *(unsigned*)&As[arow_l][akoff + 4 * q + 2] = p1.u;
            }
            if (k0 + BK < K)
                __builtin_prefetch(Arow + k0 + BK + akoff, 0, 1);
        } else {
            #pragma unroll
            for (int jj = 0; jj < 8; ++jj) {
                union { _Float16 h[2]; unsigned u; } pk;
                #pragma unroll
                for (int b = 0; b < 2; ++b) {
                    int kk = akoff + 2 * jj + b;
                    float v = 0.0f;
                    if (arow_ok && (k0 + kk) < K) v = Arow[k0 + kk];
                    pk.h[b] = (_Float16)v;
                }
                *(unsigned*)&As[arow_l][akoff + 2 * jj] = pk.u;
            }
        }

        // ---- stage B tile transposed into Bs[n][k] ----
        if (EXACT) {
            const float4* brow = (const float4*)(B + (long)(k0 + bkk) * ldb + (n0 + bnb));
            float4 v0 = brow[0], v1 = brow[1];
            Bs[bnb + 0][bkk] = (_Float16)v0.x;
            Bs[bnb + 1][bkk] = (_Float16)v0.y;
            Bs[bnb + 2][bkk] = (_Float16)v0.z;
            Bs[bnb + 3][bkk] = (_Float16)v0.w;
            Bs[bnb + 4][bkk] = (_Float16)v1.x;
            Bs[bnb + 5][bkk] = (_Float16)v1.y;
            Bs[bnb + 6][bkk] = (_Float16)v1.z;
            Bs[bnb + 7][bkk] = (_Float16)v1.w;
            if (k0 + BK < K)
                __builtin_prefetch((const float*)brow + (long)BK * ldb, 0, 1);
        } else {
            const int gk = k0 + bkk;
            if (gk < K) {
                const float* brow = B + (long)gk * ldb + (n0 + bnb);
                #pragma unroll
                for (int j = 0; j < 8; ++j) {
                    float v = ((n0 + bnb + j) < N) ? brow[j] : 0.0f;
                    Bs[bnb + j][bkk] = (_Float16)v;
                }
            } else {
                #pragma unroll
                for (int j = 0; j < 8; ++j)
                    Bs[bnb + j][bkk] = (_Float16)0.0f;
            }
        }
        __syncthreads();

        // ---- fragments (ISA 05_wmma layouts) + 4 WMMAs ----
        union Frag { v16h v; unsigned u[8]; };
        Frag af[2], bf[2];
        #pragma unroll
        for (int fm = 0; fm < 2; ++fm) {
            const _Float16* row = &As[wm * 32 + fm * 16 + l15][0];
            const int kb = hi * 8;
            #pragma unroll
            for (int j = 0; j < 8; ++j) {
                int off = (j < 4) ? (kb + 2 * j) : (kb + 2 * j + 8);
                af[fm].u[j] = *(const unsigned*)(row + off);
            }
        }
        #pragma unroll
        for (int fn = 0; fn < 2; ++fn) {
            const _Float16* row = &Bs[wn * 32 + fn * 16 + l15][0];
            const int kb = hi * 8;
            #pragma unroll
            for (int j = 0; j < 8; ++j) {
                int off = (j < 4) ? (kb + 2 * j) : (kb + 2 * j + 8);
                bf[fn].u[j] = *(const unsigned*)(row + off);
            }
        }
        #pragma unroll
        for (int fm = 0; fm < 2; ++fm)
            #pragma unroll
            for (int fn = 0; fn < 2; ++fn)
                acc[fm][fn] = __builtin_amdgcn_wmma_f32_16x16x32_f16(
                    false, af[fm].v, false, bf[fn].v,
                    (short)0, acc[fm][fn], false, false);
        __syncthreads();
    }

    // ---- epilogue: VGPR r -> row r + 8*hi ; col = l15 ----
    #pragma unroll
    for (int fm = 0; fm < 2; ++fm) {
        #pragma unroll
        for (int fn = 0; fn < 2; ++fn) {
            const int gn = n0 + wn * 32 + fn * 16 + l15;
            const bool ncol = EXACT || (gn < N);
            const float bv = (bias && ncol) ? bias[gn] : 0.0f;
            #pragma unroll
            for (int r = 0; r < 8; ++r) {
                const int gm = m0 + wm * 32 + fm * 16 + hi * 8 + r;
                if (EXACT || (gm < M && ncol)) {
                    float v = acc[fm][fn][r] + bv;
                    long off = (long)gm * ldc + gn;
                    if (beta) v += C[off];
                    if (relu) v = v > 0.0f ? v : 0.0f;
                    C[off] = v;
                }
            }
        }
    }
}

// ---------------------------------------------------------------------------
// Small helper kernels
// ---------------------------------------------------------------------------
__device__ __forceinline__ float blk_max(float v, float* red) {
    int t = threadIdx.x;
    red[t] = v; __syncthreads();
    for (int st = 128; st > 0; st >>= 1) {
        if (t < st) red[t] = fmaxf(red[t], red[t + st]);
        __syncthreads();
    }
    float r = red[0]; __syncthreads();
    return r;
}
__device__ __forceinline__ float blk_sum(float v, float* red) {
    int t = threadIdx.x;
    red[t] = v; __syncthreads();
    for (int st = 128; st > 0; st >>= 1) {
        if (t < st) red[t] += red[t + st];
        __syncthreads();
    }
    float r = red[0]; __syncthreads();
    return r;
}

__global__ void __launch_bounds__(256)
softmax_rows(const float* __restrict__ x, float* __restrict__ out, int C)
{
    __shared__ float red[256];
    const float* row = x + (long)blockIdx.x * C;
    float* orow = out + (long)blockIdx.x * C;
    float mx = -1e30f;
    for (int c = threadIdx.x; c < C; c += 256) mx = fmaxf(mx, row[c]);
    mx = blk_max(mx, red);
    float s = 0.f;
    for (int c = threadIdx.x; c < C; c += 256) s += expf(row[c] - mx);
    s = blk_sum(s, red);
    for (int c = threadIdx.x; c < C; c += 256) orow[c] = expf(row[c] - mx) / s;
}

// per-class NMS: one block (512 threads) per class c in [1, CO)
__global__ void __launch_bounds__(512)
nms_kernel(const float* __restrict__ probs0, const float* __restrict__ boxes,
           float* __restrict__ mask, int C)
{
    const int cls = blockIdx.x + 1;
    const int t = threadIdx.x;              // N == 512 == blockDim
    __shared__ unsigned long long key[512];
    __shared__ float bx1[512], by1[512], bx2[512], by2[512], ar[512];
    __shared__ int supp[512];

    // stable sort key: ascending in (-score, idx)  == descending score
    float ns = -probs0[(long)t * C + cls];
    unsigned u = __float_as_uint(ns);
    u = (u & 0x80000000u) ? ~u : (u | 0x80000000u);
    key[t] = ((unsigned long long)u << 32) | (unsigned)t;
    __syncthreads();

    for (int k = 2; k <= 512; k <<= 1)
        for (int j = k >> 1; j > 0; j >>= 1) {
            int ixj = t ^ j;
            if (ixj > t) {
                bool up = ((t & k) == 0);
                unsigned long long a = key[t], b = key[ixj];
                if ((a > b) == up) { key[t] = b; key[ixj] = a; }
            }
            __syncthreads();
        }

    const int oi = (int)(key[t] & 0xFFFFFFFFu);
    const float* bb = boxes + ((long)oi * C + cls) * 4;
    float x1 = bb[0], y1 = bb[1], x2 = bb[2], y2 = bb[3];
    bx1[t] = x1; by1[t] = y1; bx2[t] = x2; by2[t] = y2;
    ar[t] = fmaxf(x2 - x1, 0.f) * fmaxf(y2 - y1, 0.f);
    supp[t] = 0;
    __syncthreads();

    for (int i = 0; i < 511; ++i) {
        if (!supp[i] && t > i) {
            float xx1 = fmaxf(bx1[i], x1), yy1 = fmaxf(by1[i], y1);
            float xx2 = fminf(bx2[i], x2), yy2 = fminf(by2[i], y2);
            float inter = fmaxf(xx2 - xx1, 0.f) * fmaxf(yy2 - yy1, 0.f);
            float iou = inter / (ar[i] + ar[t] - inter + 1e-8f);
            if (iou > 0.3f) supp[t] = 1;
        }
        __syncthreads();
    }
    mask[(long)oi * C + cls] = supp[t] ? 0.f : 1.f;
}

__global__ void __launch_bounds__(256)
masked_softmax_pred(const float* __restrict__ logits, const float* __restrict__ mask,
                    const float* __restrict__ probs0, float* __restrict__ p,
                    float* __restrict__ out_logits, float* __restrict__ out_preds, int C)
{
    __shared__ float red[256];
    __shared__ int   redi[256];
    const int row = blockIdx.x, t = threadIdx.x;
    const float* x  = logits + (long)row * C;
    const float* mk = mask   + (long)row * C;

    for (int c = t; c < C; c += 256) out_logits[(long)row * C + c] = x[c];

    float mx = -1e30f;
    for (int c = t; c < C; c += 256) {
        float lp = mk[c] * x[c] + (1.f - mk[c]) * (-1000.f);
        mx = fmaxf(mx, lp);
    }
    mx = blk_max(mx, red);
    float s = 0.f;
    for (int c = t; c < C; c += 256) {
        float lp = mk[c] * x[c] + (1.f - mk[c]) * (-1000.f);
        s += expf(lp - mx);
    }
    s = blk_sum(s, red);
    for (int c = t; c < C; c += 256) {
        float lp = mk[c] * x[c] + (1.f - mk[c]) * (-1000.f);
        p[(long)row * C + c] = expf(lp - mx) / s;
    }

    // argmax((mask*probs0)[:,1:]) + 1 ; first-occurrence tie break
    float best = -1e30f; int bi = 0x7fffffff;
    for (int c = 1 + t; c < C; c += 256) {
        float v = mk[c] * probs0[(long)row * C + c];
        if (v > best) { best = v; bi = c; }
    }
    red[t] = best; redi[t] = bi; __syncthreads();
    for (int st = 128; st > 0; st >>= 1) {
        if (t < st) {
            if (red[t + st] > red[t] ||
                (red[t + st] == red[t] && redi[t + st] < redi[t])) {
                red[t] = red[t + st]; redi[t] = redi[t + st];
            }
        }
        __syncthreads();
    }
    if (t == 0) out_preds[row] = (float)redi[0];
}

__global__ void zero_f(float* __restrict__ p, int n) {
    int i = blockIdx.x * 256 + threadIdx.x;
    if (i < n) p[i] = 0.f;
}
__global__ void extract_so(const int* __restrict__ rel, int* __restrict__ s,
                           int* __restrict__ o, int E) {
    int e = blockIdx.x * 256 + threadIdx.x;
    if (e < E) { s[e] = rel[3 * e + 1]; o[e] = rel[3 * e + 2]; }
}
__global__ void deg_scatter(const int* __restrict__ s, const int* __restrict__ o,
                            float* __restrict__ deg, int E) {
    int e = blockIdx.x * 256 + threadIdx.x;
    if (e < E) { atomicAdd(&deg[s[e]], 1.f); atomicAdd(&deg[o[e]], 1.f); }
}
__global__ void edge_w_kernel(const float* __restrict__ pK, const float* __restrict__ p,
                              const int* __restrict__ s, const int* __restrict__ o,
                              float* __restrict__ w, int E, int C) {
    int e = blockIdx.x * 128 + threadIdx.x;
    if (e < E) {
        const float* a = pK + (long)s[e] * C;
        const float* b = p  + (long)o[e] * C;
        float acc = 0.f;
        for (int c = 0; c < C; ++c) acc += a[c] * b[c];
        w[e] = acc;
    }
}
__global__ void scatter_msg(const float* __restrict__ t1, const float* __restrict__ t2,
                            const float* __restrict__ w, const int* __restrict__ s,
                            const int* __restrict__ o, float* __restrict__ m,
                            int E, int H) {
    int idx = blockIdx.x * 256 + threadIdx.x;
    if (idx < E * H) {
        int e = idx / H, j = idx - e * H;
        float we = w[e];
        atomicAdd(&m[(long)o[e] * H + j], we * t1[idx]);
        atomicAdd(&m[(long)s[e] * H + j], we * t2[idx]);
    }
}
__global__ void mdiv_kernel(float* __restrict__ m, const float* __restrict__ deg,
                            int n, int H) {
    int i = blockIdx.x * 256 + threadIdx.x;
    if (i < n) m[i] /= fmaxf(deg[i / H], 1.f);
}
__global__ void gru_zr(float* __restrict__ zb, float* __restrict__ rb,
                       const float* __restrict__ h, int n) {
    int i = blockIdx.x * 256 + threadIdx.x;
    if (i < n) {
        float z = 1.f / (1.f + expf(-zb[i])); zb[i] = z;
        float r = 1.f / (1.f + expf(-rb[i])); rb[i] = r * h[i];
    }
}
__global__ void gru_upd(float* __restrict__ h, const float* __restrict__ z,
                        const float* __restrict__ hb, int n) {
    int i = blockIdx.x * 256 + threadIdx.x;
    if (i < n) {
        float hh = tanhf(hb[i]);
        float zz = z[i];
        h[i] = (1.f - zz) * h[i] + zz * hh;
    }
}

// ---------------------------------------------------------------------------
static inline void launch_gemm(hipStream_t st, const float* A, int lda,
                               const float* B, int ldb, float* C, int ldc,
                               int M, int N, int K, const float* bias,
                               const int* gather, int beta, int relu) {
    dim3 grid((N + BN - 1) / BN, (M + BM - 1) / BM);
    const bool exact = (M % BM == 0) && (N % BN == 0) && (K % BK == 0) &&
                       ((lda & 3) == 0) && ((ldb & 3) == 0) &&
                       ((((size_t)A) & 15) == 0) && ((((size_t)B) & 15) == 0);
    if (exact)
        gemm_wmma_kernel<true><<<grid, 256, 0, st>>>(A, lda, B, ldb, C, ldc,
                                                     M, N, K, bias, gather, beta, relu);
    else
        gemm_wmma_kernel<false><<<grid, 256, 0, st>>>(A, lda, B, ldb, C, ldc,
                                                      M, N, K, bias, gather, beta, relu);
}

extern "C" void kernel_launch(void* const* d_in, const int* in_sizes, int n_in,
                              void* d_out, int out_size, void* d_ws, size_t ws_size,
                              hipStream_t stream) {
    (void)in_sizes; (void)n_in; (void)out_size; (void)ws_size;
    const float* obj_fmaps  = (const float*)d_in[1];
    const float* obj_logits = (const float*)d_in[2];
    const int*   rel_inds   = (const int*)d_in[3];
    const float* vr         = (const float*)d_in[4];
    const float* boxes      = (const float*)d_in[5];
    const float* Wop = (const float*)d_in[6],  *bop = (const float*)d_in[7];
    const float* Wrp = (const float*)d_in[8],  *brp = (const float*)d_in[9];
    const float* Emb = (const float*)d_in[10], *Wemb = (const float*)d_in[11];
    const float* Km  = (const float*)d_in[12];
    const float* Wf  = (const float*)d_in[13], *Uf = (const float*)d_in[14];
    const float* Wb  = (const float*)d_in[15], *Ub = (const float*)d_in[16];
    const float* Wz  = (const float*)d_in[17], *Uz = (const float*)d_in[18], *bz = (const float*)d_in[19];
    const float* Wr  = (const float*)d_in[20], *Ur = (const float*)d_in[21], *br = (const float*)d_in[22];
    const float* Wh  = (const float*)d_in[23], *Uh = (const float*)d_in[24], *bh = (const float*)d_in[25];
    const float* W1  = (const float*)d_in[26], *b1 = (const float*)d_in[27];
    const float* W2  = (const float*)d_in[28], *b2 = (const float*)d_in[29];

    constexpr int N = 512, E = 4096, OD = 4096, H = 512, CO = 151, CR = 51, EM = 300;

    char* base = (char*)d_ws;
    size_t off = 0;
    auto alloc = [&](size_t nbytes) -> void* {
        void* pp = base + off;
        off += (nbytes + 255) & ~(size_t)255;
        return pp;
    };
    float* probs0 = (float*)alloc((size_t)N * CO * 4);
    float* mask   = (float*)alloc((size_t)N * CO * 4);
    float* p      = (float*)alloc((size_t)N * CO * 4);
    float* pK     = (float*)alloc((size_t)N * CO * 4);
    float* pe     = (float*)alloc((size_t)N * EM * 4);
    float* h      = (float*)alloc((size_t)N * H * 4);
    float* f      = (float*)alloc((size_t)E * H * 4);
    float* wbuf   = (float*)alloc((size_t)E * 4);
    float* deg    = (float*)alloc((size_t)N * 4);
    int*   sbuf   = (int*)alloc((size_t)E * 4);
    int*   obuf   = (int*)alloc((size_t)E * 4);
    float* t1     = (float*)alloc((size_t)E * H * 4);
    float* t2     = (float*)alloc((size_t)E * H * 4);
    float* m      = (float*)alloc((size_t)N * H * 4);
    float* zb     = (float*)alloc((size_t)N * H * 4);
    float* rb     = (float*)alloc((size_t)N * H * 4);
    float* hb     = (float*)alloc((size_t)N * H * 4);
    float* X      = (float*)alloc((size_t)E * H * 4);

    float* out_logits = (float*)d_out;
    float* out_preds  = out_logits + (size_t)N * CO;
    float* out_rel    = out_preds + N;

    softmax_rows<<<N, 256, 0, stream>>>(obj_logits, probs0, CO);
    zero_f<<<(N * CO + 255) / 256, 256, 0, stream>>>(mask, N * CO);
    nms_kernel<<<CO - 1, 512, 0, stream>>>(probs0, boxes, mask, CO);
    masked_softmax_pred<<<N, 256, 0, stream>>>(obj_logits, mask, probs0, p,
                                               out_logits, out_preds, CO);
    extract_so<<<(E + 255) / 256, 256, 0, stream>>>(rel_inds, sbuf, obuf, E);

    // h = obj_fmaps@Wop + bop + (p@Emb)@Wemb
    launch_gemm(stream, obj_fmaps, OD, Wop, H, h, H, N, H, OD, bop, nullptr, 0, 0);
    launch_gemm(stream, p, CO, Emb, EM, pe, EM, N, EM, CO, nullptr, nullptr, 0, 0);
    launch_gemm(stream, pe, EM, Wemb, H, h, H, N, H, EM, nullptr, nullptr, 1, 0);
    // f = vr@Wrp + brp
    launch_gemm(stream, vr, OD, Wrp, H, f, H, E, H, OD, brp, nullptr, 0, 0);
    // w[e] = <(p@K)[s], p[o]>
    launch_gemm(stream, p, CO, Km, CO, pK, CO, N, CO, CO, nullptr, nullptr, 0, 0);
    edge_w_kernel<<<(E + 127) / 128, 128, 0, stream>>>(pK, p, sbuf, obuf, wbuf, E, CO);
    zero_f<<<(N + 255) / 256, 256, 0, stream>>>(deg, N);
    deg_scatter<<<(E + 255) / 256, 256, 0, stream>>>(sbuf, obuf, deg, E);

    for (int step = 0; step < 3; ++step) {
        launch_gemm(stream, h, H, Wf, H, t1, H, E, H, H, nullptr, sbuf, 0, 0);
        launch_gemm(stream, f, H, Uf, H, t1, H, E, H, H, nullptr, nullptr, 1, 0);
        launch_gemm(stream, h, H, Wb, H, t2, H, E, H, H, nullptr, obuf, 0, 0);
        launch_gemm(stream, f, H, Ub, H, t2, H, E, H, H, nullptr, nullptr, 1, 0);
        zero_f<<<(N * H + 255) / 256, 256, 0, stream>>>(m, N * H);
        scatter_msg<<<(E * H + 255) / 256, 256, 0, stream>>>(t1, t2, wbuf, sbuf, obuf, m, E, H);
        mdiv_kernel<<<(N * H + 255) / 256, 256, 0, stream>>>(m, deg, N * H, H);
        launch_gemm(stream, m, H, Wz, H, zb, H, N, H, H, nullptr, nullptr, 0, 0);
        launch_gemm(stream, h, H, Uz, H, zb, H, N, H, H, bz, nullptr, 1, 0);
        launch_gemm(stream, m, H, Wr, H, rb, H, N, H, H, nullptr, nullptr, 0, 0);
        launch_gemm(stream, h, H, Ur, H, rb, H, N, H, H, br, nullptr, 1, 0);
        gru_zr<<<(N * H + 255) / 256, 256, 0, stream>>>(zb, rb, h, N * H);
        launch_gemm(stream, m, H, Wh, H, hb, H, N, H, H, nullptr, nullptr, 0, 0);
        launch_gemm(stream, rb, H, Uh, H, hb, H, N, H, H, bh, nullptr, 1, 0);
        gru_upd<<<(N * H + 255) / 256, 256, 0, stream>>>(h, zb, hb, N * H);
    }

    // rel_logits = relu([h[s], h[o], f] @ W1 + b1) @ W2 + b2
    launch_gemm(stream, h, H, W1,                   H, X, H, E, H, H, nullptr, sbuf, 0, 0);
    launch_gemm(stream, h, H, W1 + (size_t)H * H,   H, X, H, E, H, H, nullptr, obuf, 1, 0);
    launch_gemm(stream, f, H, W1 + (size_t)2*H * H, H, X, H, E, H, H, b1, nullptr, 1, 1);
    launch_gemm(stream, X, H, W2, CR, out_rel, CR, E, CR, H, b2, nullptr, 0, 0);
}
